// MyLeNetRotateInvariant_35845797052768
// MI455X (gfx1250) — matrix operand order repaired and checked by reading, compiled, not run on hardware
//
#include <hip/hip_runtime.h>

// ---------- types ----------
typedef __attribute__((ext_vector_type(16))) _Float16     v16h;
typedef __attribute__((ext_vector_type(8)))  float        v8f;
typedef __attribute__((ext_vector_type(4)))  float        vf4;
typedef __attribute__((ext_vector_type(2)))  unsigned int vu2;
typedef __attribute__((ext_vector_type(2)))  int          v2i;

union AFrag { v16h v; vf4 f[2]; };

// Async global->LDS staging (CDNA5): use if the toolchain exposes the builtins.
#if __has_builtin(__builtin_amdgcn_global_load_async_to_lds_b64) && \
    __has_builtin(__builtin_amdgcn_s_wait_asynccnt)
#define USE_ASYNC_LDS 1
#else
#define USE_ASYNC_LDS 0
#endif

#define AS_GLOBAL __attribute__((address_space(1)))
#define AS_LDS    __attribute__((address_space(3)))

// ---------- rotation pad tables (left, top) from reference L ----------
__device__ __constant__ int cLL[8][3] = {
  {0,1,2},{1,1,1},{2,1,0},{2,1,0},{2,1,0},{1,1,1},{0,1,2},{0,1,2}};
__device__ __constant__ int cLT[8][3] = {
  {2,1,0},{2,1,1},{2,1,0},{1,1,1},{0,1,2},{0,1,2},{0,1,2},{1,1,1}};

// ---------- workspace layout (bytes) ----------
static constexpr size_t OFF_K1 = 0;                    // 48*25 f32      (4800 B)
static constexpr size_t OFF_K2 = 8192;                 // 25*2*128*32 f16 (409600 B)
static constexpr size_t OFF_K3 = 417792;               // 25*4*80*32 f16  (512000 B)
static constexpr size_t OFF_ST = 929792;               // 88 f32 stats
static constexpr size_t OFF_A1 = 933888;               // act1p f16 [256][32][32][64]
static constexpr size_t OFF_A2 = 34488320;             // act2p f16 [256][16][16][128]
static constexpr size_t OFF_A3 = 51265536;             // act3  f32 [256][80][16][16]

// stats float indices: 0 sum1(6) | 6 sq1(6) | 12 scale1(6) | 18 shift1(6)
//                      24 sum2(16) | 40 sq2(16) | 56 scale2(16) | 72 shift2(16)

// ===================================================================
// Build effective 5x5 kernels (fold double-pads into tap offsets)
// ===================================================================
__global__ void build_k1(const float* __restrict__ w1, float* __restrict__ K1) {
  int gid = threadIdx.x;
  if (gid >= 48) return;
  int i = gid / 6, oc = gid % 6;
  int i2 = (i + 4) & 7;
  float k5[25];
#pragma unroll
  for (int t = 0; t < 25; ++t) k5[t] = 0.f;
#pragma unroll
  for (int j = 0; j < 3; ++j)
#pragma unroll
    for (int k = 0; k < 3; ++k) {
      int dy = cLT[i][j] + cLT[i2][k];
      int dx = cLL[i][j] + cLL[i2][k];
      k5[(4 - dy) * 5 + (4 - dx)] += w1[oc * 9 + j * 3 + k];
    }
#pragma unroll
  for (int t = 0; t < 25; ++t) K1[gid * 25 + t] = k5[t];
}

// K2 layout: [tap 25][kb 2][n 128][kk 32] f16 ; n = rot*16 + oc ; in = kb*32+kk
__global__ void build_k2(const float* __restrict__ w2, _Float16* __restrict__ K2) {
  int gid = blockIdx.x * 256 + threadIdx.x;
  if (gid >= 8 * 16 * 64) return;
  int ic = gid & 63, oc = (gid >> 6) & 15, i = gid >> 10;
  float w[3][3];
  if (ic < 48) {
    int r2 = ic / 6, gi = ic % 6;
    int rs = ((r2 - i) % 8 + 8) % 8;
#pragma unroll
    for (int j = 0; j < 3; ++j)
#pragma unroll
      for (int k = 0; k < 3; ++k)
        w[j][k] = w2[(oc * 48 + rs * 6 + gi) * 9 + j * 3 + k];
  } else {
#pragma unroll
    for (int j = 0; j < 3; ++j)
#pragma unroll
      for (int k = 0; k < 3; ++k) w[j][k] = 0.f;
  }
  float k5[25];
#pragma unroll
  for (int t = 0; t < 25; ++t) k5[t] = 0.f;
  int i2 = (i + 4) & 7;
#pragma unroll
  for (int j = 0; j < 3; ++j)
#pragma unroll
    for (int k = 0; k < 3; ++k) {
      int dy = cLT[i][j] + cLT[i2][k];
      int dx = cLL[i][j] + cLL[i2][k];
      k5[(4 - dy) * 5 + (4 - dx)] += w[j][k];
    }
  int n = i * 16 + oc, kb = ic >> 5, kk = ic & 31;
#pragma unroll
  for (int t = 0; t < 25; ++t)
    K2[((t * 2 + kb) * 128 + n) * 32 + kk] = (_Float16)k5[t];
}

// K3 layout: [tap 25][kb 4][n 80][kk 32] f16 ; n = rot*10 + oc
__global__ void build_k3(const float* __restrict__ w3, _Float16* __restrict__ K3) {
  int gid = blockIdx.x * 256 + threadIdx.x;
  if (gid >= 8 * 10 * 128) return;
  int ic = gid & 127, oc = (gid >> 7) % 10, i = gid / 1280;
  int r2 = ic >> 4, gi = ic & 15;
  int rs = ((r2 - i) % 8 + 8) % 8;
  float w[3][3];
#pragma unroll
  for (int j = 0; j < 3; ++j)
#pragma unroll
    for (int k = 0; k < 3; ++k)
      w[j][k] = w3[(oc * 128 + rs * 16 + gi) * 9 + j * 3 + k];
  float k5[25];
#pragma unroll
  for (int t = 0; t < 25; ++t) k5[t] = 0.f;
  int i2 = (i + 4) & 7;
#pragma unroll
  for (int j = 0; j < 3; ++j)
#pragma unroll
    for (int k = 0; k < 3; ++k) {
      int dy = cLT[i][j] + cLT[i2][k];
      int dx = cLL[i][j] + cLL[i2][k];
      k5[(4 - dy) * 5 + (4 - dx)] += w[j][k];
    }
  int n = i * 10 + oc, kb = ic >> 5, kk = ic & 31;
#pragma unroll
  for (int t = 0; t < 25; ++t)
    K3[((t * 4 + kb) * 80 + n) * 32 + kk] = (_Float16)k5[t];
}

// ===================================================================
// Layer 1 (in_c = 1): scalar conv; pass A = BN stats, pass B = BN+ReLU+pool
// ===================================================================
__global__ __launch_bounds__(256) void conv1_stats(
    const float* __restrict__ x, const float* __restrict__ K1, float* __restrict__ st) {
  __shared__ float sK1[1200];
  __shared__ float ssum[6], ssq[6];
  int tid = threadIdx.x;
  for (int i = tid; i < 1200; i += 256) sK1[i] = K1[i];
  if (tid < 6) { ssum[tid] = 0.f; ssq[tid] = 0.f; }
  __syncthreads();

  int gid = blockIdx.x * 256 + tid;
  int b = gid >> 12, pix = gid & 4095;
  int y = pix >> 6, xx = pix & 63;
  const float* xb = x + b * 4096;
  float p[25];
#pragma unroll
  for (int dy = 0; dy < 5; ++dy)
#pragma unroll
    for (int dx = 0; dx < 5; ++dx) {
      int yy = y + dy - 2, xc = xx + dx - 2;
      p[dy * 5 + dx] = (yy >= 0 && yy < 64 && xc >= 0 && xc < 64) ? xb[yy * 64 + xc] : 0.f;
    }
  float lsum[6], lsq[6];
#pragma unroll
  for (int c = 0; c < 6; ++c) { lsum[c] = 0.f; lsq[c] = 0.f; }
  for (int cr = 0; cr < 48; ++cr) {
    const float* kk = &sK1[cr * 25];
    float a = 0.f;
#pragma unroll
    for (int t = 0; t < 25; ++t) a += kk[t] * p[t];
    int c = cr % 6;
    lsum[c] += a; lsq[c] += a * a;
  }
#pragma unroll
  for (int c = 0; c < 6; ++c) { atomicAdd(&ssum[c], lsum[c]); atomicAdd(&ssq[c], lsq[c]); }
  __syncthreads();
  if (tid < 6)  atomicAdd(&st[tid], ssum[tid]);
  if (tid >= 6 && tid < 12) atomicAdd(&st[tid], ssq[tid - 6]);
}

__global__ void finalize_bn(const float* __restrict__ sum, const float* __restrict__ sq,
                            const float* __restrict__ gamma, const float* __restrict__ beta,
                            float* __restrict__ scale, float* __restrict__ shift,
                            int C, float invN) {
  int c = threadIdx.x;
  if (c < C) {
    float mean = sum[c] * invN;
    float var  = sq[c] * invN - mean * mean;
    float sc   = gamma[c] * rsqrtf(var + 1e-5f);
    scale[c] = sc;
    shift[c] = beta[c] - mean * sc;
  }
}

// one thread per pooled output pixel; writes act1p NHWC f16 [b][32][32][64]
__global__ __launch_bounds__(256) void conv1_bnpool(
    const float* __restrict__ x, const float* __restrict__ K1,
    const float* __restrict__ scale1, const float* __restrict__ shift1,
    _Float16* __restrict__ a1) {
  __shared__ float sK1[1200];
  __shared__ float ssc[6], ssh[6];
  int tid = threadIdx.x;
  for (int i = tid; i < 1200; i += 256) sK1[i] = K1[i];
  if (tid < 6) { ssc[tid] = scale1[tid]; ssh[tid] = shift1[tid]; }
  __syncthreads();

  int gid = blockIdx.x * 256 + tid;
  int b = gid >> 10, pix = gid & 1023;
  int py = pix >> 5, px = pix & 31;
  const float* xb = x + b * 4096;
  float p[6][6];
#pragma unroll
  for (int dy = 0; dy < 6; ++dy)
#pragma unroll
    for (int dx = 0; dx < 6; ++dx) {
      int yy = 2 * py + dy - 2, xc = 2 * px + dx - 2;
      p[dy][dx] = (yy >= 0 && yy < 64 && xc >= 0 && xc < 64) ? xb[yy * 64 + xc] : 0.f;
    }
  _Float16* dst = &a1[((b * 32 + py) * 32 + px) * 64];
  for (int cr = 0; cr < 48; ++cr) {
    const float* kk = &sK1[cr * 25];
    float sc = ssc[cr % 6], sh = ssh[cr % 6];
    float m = 0.f;
#pragma unroll
    for (int dy = 0; dy < 2; ++dy)
#pragma unroll
      for (int dx = 0; dx < 2; ++dx) {
        float a = 0.f;
#pragma unroll
        for (int ty = 0; ty < 5; ++ty)
#pragma unroll
          for (int tx = 0; tx < 5; ++tx)
            a += kk[ty * 5 + tx] * p[dy + ty][dx + tx];
        float v = fmaxf(a * sc + sh, 0.f);
        m = fmaxf(m, v);
      }
    dst[cr] = (_Float16)m;
  }
}

// ===================================================================
// Layer 2: WMMA implicit GEMM. Block = 8 waves; wave = 16 px (2x8) x N=128.
// STATS pass accumulates BN sums; apply pass fuses BN+ReLU+2x2pool.
// ===================================================================
template <bool STATS>
__global__ __launch_bounds__(256) void conv2_kernel(
    const _Float16* __restrict__ a1, const _Float16* __restrict__ K2,
    float* __restrict__ gsum, float* __restrict__ gsq,
    const float* __restrict__ scale2, const float* __restrict__ shift2,
    _Float16* __restrict__ a2) {
  __shared__ _Float16 lds[20 * 12 * 64];
  __shared__ float ssum[16], ssq[16];
  int tid = threadIdx.x;
  int blk = blockIdx.x;
  int b = blk >> 3, r = blk & 7;
  int y0 = (r >> 2) * 16, x0 = (r & 3) * 8;

  // stage input patch [20 rows][12 cols][64 ch] into LDS (NHWC, 8B chunks)
#pragma unroll
  for (int it = 0; it < 15; ++it) {
    int idx = it * 256 + tid;           // 3840 chunks of 4 f16
    int ch4 = idx & 15;
    int col = (idx >> 4) % 12;
    int row = idx / 192;
    int gy = y0 - 2 + row, gx = x0 - 2 + col;
    _Float16* lp = &lds[((row * 12 + col) << 6) + (ch4 << 2)];
    bool inb = (gy >= 0 && gy < 32 && gx >= 0 && gx < 32);
#if USE_ASYNC_LDS
    if (inb) {
      const _Float16* gp = &a1[(((b * 32 + gy) * 32 + gx) << 6) + (ch4 << 2)];
      __builtin_amdgcn_global_load_async_to_lds_b64(
          (AS_GLOBAL v2i*)(AS_GLOBAL void*)gp, (AS_LDS v2i*)(AS_LDS void*)lp, 0, 0);
    } else {
      *(vu2*)lp = (vu2)0u;
    }
#else
    vu2 v = (vu2)0u;
    if (inb)
      v = *(const vu2*)&a1[(((b * 32 + gy) * 32 + gx) << 6) + (ch4 << 2)];
    *(vu2*)lp = v;
#endif
  }
  if (STATS && tid < 16) { ssum[tid] = 0.f; ssq[tid] = 0.f; }
#if USE_ASYNC_LDS
  __builtin_amdgcn_s_wait_asynccnt(0);
#endif
  __syncthreads();

  int w = tid >> 5, lane = tid & 31;
  int m = lane & 15, h = lane >> 4;
  int ady = m >> 3, adx = m & 7;

  v8f acc[8];
#pragma unroll
  for (int nt = 0; nt < 8; ++nt) acc[nt] = (v8f){0.f,0.f,0.f,0.f,0.f,0.f,0.f,0.f};

#pragma unroll 1
  for (int t = 0; t < 25; ++t) {
    int ty = t / 5, tx = t % 5;
    if (t < 24) __builtin_prefetch(&K2[(size_t)((t + 1) * 2) * 128 * 32], 0, 0);
    const _Float16* abase = &lds[(((2 * w + ady + ty) * 12 + (adx + tx)) << 6)];
#pragma unroll
    for (int kb = 0; kb < 2; ++kb) {
      AFrag a;
      a.f[0] = *(const vf4*)(abase + kb * 32 + 8 * h);
      a.f[1] = *(const vf4*)(abase + kb * 32 + 16 + 8 * h);
      const _Float16* bbase = &K2[(size_t)(((t * 2 + kb) * 128 + (lane & 15)) * 32 + h * 8)];
#pragma unroll
      for (int nt = 0; nt < 8; ++nt) {
        AFrag bf;
        bf.f[0] = *(const vf4*)(bbase + nt * 512);
        bf.f[1] = *(const vf4*)(bbase + nt * 512 + 16);
        acc[nt] = __builtin_amdgcn_wmma_f32_16x16x32_f16(
            false, a.v, false, bf.v, (short)0, acc[nt], false, false);
      }
    }
  }

  if (STATS) {
    float ls = 0.f, lq = 0.f;
#pragma unroll
    for (int nt = 0; nt < 8; ++nt)
#pragma unroll
      for (int v = 0; v < 8; ++v) { float xv = acc[nt][v]; ls += xv; lq += xv * xv; }
    atomicAdd(&ssum[lane & 15], ls);
    atomicAdd(&ssq[lane & 15], lq);
    __syncthreads();
    if (tid < 16) { atomicAdd(&gsum[tid], ssum[tid]); atomicAdd(&gsq[tid], ssq[tid]); }
  } else {
    float sc = scale2[lane & 15], sh = shift2[lane & 15];
    int py = (y0 >> 1) + w, px0 = x0 >> 1;
#pragma unroll
    for (int nt = 0; nt < 8; ++nt) {
      float rmax[8];
#pragma unroll
      for (int v = 0; v < 8; ++v) {
        float tv = fmaxf(acc[nt][v] * sc + sh, 0.f);
        rmax[v] = fmaxf(tv, __shfl_xor(tv, 16, 32));   // max over pool rows
      }
      if (lane < 16) {
#pragma unroll
        for (int u = 0; u < 4; ++u) {
          float pv = fmaxf(rmax[2 * u], rmax[2 * u + 1]);  // max over pool cols
          a2[(((b * 16 + py) * 16 + px0 + u) << 7) + nt * 16 + lane] = (_Float16)pv;
        }
      }
    }
  }
}

// ===================================================================
// Layer 3: WMMA implicit GEMM, N=80, K = 4x32 per tap; raw f32 out.
// ===================================================================
__global__ __launch_bounds__(256) void conv3_kernel(
    const _Float16* __restrict__ a2, const _Float16* __restrict__ K3,
    float* __restrict__ a3) {
  __shared__ _Float16 lds[20 * 12 * 128];
  int tid = threadIdx.x;
  int blk = blockIdx.x;
  int b = blk >> 1;
  int x0 = (blk & 1) * 8;

#pragma unroll
  for (int it = 0; it < 30; ++it) {
    int idx = it * 256 + tid;           // 7680 chunks of 4 f16
    int ch4 = idx & 31;
    int col = (idx >> 5) % 12;
    int row = idx / 384;
    int gy = row - 2, gx = x0 - 2 + col;
    _Float16* lp = &lds[((row * 12 + col) << 7) + (ch4 << 2)];
    bool inb = (gy >= 0 && gy < 16 && gx >= 0 && gx < 16);
#if USE_ASYNC_LDS
    if (inb) {
      const _Float16* gp = &a2[(((b * 16 + gy) * 16 + gx) << 7) + (ch4 << 2)];
      __builtin_amdgcn_global_load_async_to_lds_b64(
          (AS_GLOBAL v2i*)(AS_GLOBAL void*)gp, (AS_LDS v2i*)(AS_LDS void*)lp, 0, 0);
    } else {
      *(vu2*)lp = (vu2)0u;
    }
#else
    vu2 v = (vu2)0u;
    if (inb)
      v = *(const vu2*)&a2[(((b * 16 + gy) * 16 + gx) << 7) + (ch4 << 2)];
    *(vu2*)lp = v;
#endif
  }
#if USE_ASYNC_LDS
  __builtin_amdgcn_s_wait_asynccnt(0);
#endif
  __syncthreads();

  int w = tid >> 5, lane = tid & 31;
  int m = lane & 15, h = lane >> 4;
  int ady = m >> 3, adx = m & 7;

  v8f acc[5];
#pragma unroll
  for (int nt = 0; nt < 5; ++nt) acc[nt] = (v8f){0.f,0.f,0.f,0.f,0.f,0.f,0.f,0.f};

#pragma unroll 1
  for (int t = 0; t < 25; ++t) {
    int ty = t / 5, tx = t % 5;
    if (t < 24) __builtin_prefetch(&K3[(size_t)((t + 1) * 4) * 80 * 32], 0, 0);
    const _Float16* abase = &lds[(((2 * w + ady + ty) * 12 + (adx + tx)) << 7)];
#pragma unroll
    for (int kb = 0; kb < 4; ++kb) {
      AFrag a;
      a.f[0] = *(const vf4*)(abase + kb * 32 + 8 * h);
      a.f[1] = *(const vf4*)(abase + kb * 32 + 16 + 8 * h);
      const _Float16* bbase = &K3[(size_t)(((t * 4 + kb) * 80 + (lane & 15)) * 32 + h * 8)];
#pragma unroll
      for (int nt = 0; nt < 5; ++nt) {
        AFrag bf;
        bf.f[0] = *(const vf4*)(bbase + nt * 512);
        bf.f[1] = *(const vf4*)(bbase + nt * 512 + 16);
        acc[nt] = __builtin_amdgcn_wmma_f32_16x16x32_f16(
            false, a.v, false, bf.v, (short)0, acc[nt], false, false);
      }
    }
  }

  // store raw conv3: act3[b][n][y][x], y = 2w + h, x = x0 + v (contiguous per lane)
#pragma unroll
  for (int nt = 0; nt < 5; ++nt) {
    int n = nt * 16 + (lane & 15);
    int y = 2 * w + h;
    vf4 lo = {acc[nt][0], acc[nt][1], acc[nt][2], acc[nt][3]};
    vf4 hi = {acc[nt][4], acc[nt][5], acc[nt][6], acc[nt][7]};
    float* dst = &a3[(((b * 80 + n) * 16 + y) << 4) + x0];
    *(vf4*)dst = lo;
    *(vf4*)(dst + 4) = hi;
  }
}

// rotateMax (contiguous groups of 8 channels) + global spatial max
__global__ __launch_bounds__(256) void final_reduce(
    const float* __restrict__ a3, float* __restrict__ out) {
  __shared__ float red[256];
  int b = blockIdx.x / 10, cls = blockIdx.x % 10;
  const float* base = a3 + (size_t)(b * 80 + cls * 8) * 256;
  float m = -INFINITY;
  for (int i = threadIdx.x; i < 2048; i += 256) m = fmaxf(m, base[i]);
  red[threadIdx.x] = m;
  __syncthreads();
  for (int s = 128; s > 0; s >>= 1) {
    if (threadIdx.x < s) red[threadIdx.x] = fmaxf(red[threadIdx.x], red[threadIdx.x + s]);
    __syncthreads();
  }
  if (threadIdx.x == 0) out[b * 10 + cls] = red[0];
}

// ===================================================================
extern "C" void kernel_launch(void* const* d_in, const int* in_sizes, int n_in,
                              void* d_out, int out_size, void* d_ws, size_t ws_size,
                              hipStream_t stream) {
  (void)in_sizes; (void)n_in; (void)out_size; (void)ws_size;
  const float* x  = (const float*)d_in[0];
  const float* w1 = (const float*)d_in[1];
  const float* g1 = (const float*)d_in[2];
  const float* b1 = (const float*)d_in[3];
  const float* w2 = (const float*)d_in[4];
  const float* g2 = (const float*)d_in[5];
  const float* b2 = (const float*)d_in[6];
  const float* w3 = (const float*)d_in[7];
  float* out = (float*)d_out;

  char* ws = (char*)d_ws;
  float*     K1 = (float*)(ws + OFF_K1);
  _Float16*  K2 = (_Float16*)(ws + OFF_K2);
  _Float16*  K3 = (_Float16*)(ws + OFF_K3);
  float*     st = (float*)(ws + OFF_ST);
  _Float16*  a1 = (_Float16*)(ws + OFF_A1);
  _Float16*  a2 = (_Float16*)(ws + OFF_A2);
  float*     a3 = (float*)(ws + OFF_A3);

  (void)hipMemsetAsync(st, 0, 4096, stream);
  (void)hipMemsetAsync(a1, 0, (size_t)256 * 32 * 32 * 64 * sizeof(_Float16), stream);

  build_k1<<<1, 64, 0, stream>>>(w1, K1);
  build_k2<<<32, 256, 0, stream>>>(w2, K2);
  build_k3<<<40, 256, 0, stream>>>(w3, K3);

  conv1_stats<<<4096, 256, 0, stream>>>(x, K1, st);
  finalize_bn<<<1, 32, 0, stream>>>(st + 0, st + 6, g1, b1, st + 12, st + 18,
                                    6, 1.f / 8388608.f);
  conv1_bnpool<<<1024, 256, 0, stream>>>(x, K1, st + 12, st + 18, a1);

  conv2_kernel<true><<<2048, 256, 0, stream>>>(a1, K2, st + 24, st + 40,
                                               st + 56, st + 72, a2);
  finalize_bn<<<1, 32, 0, stream>>>(st + 24, st + 40, g2, b2, st + 56, st + 72,
                                    16, 1.f / 2097152.f);
  conv2_kernel<false><<<2048, 256, 0, stream>>>(a1, K2, st + 24, st + 40,
                                                st + 56, st + 72, a2);

  conv3_kernel<<<512, 256, 0, stream>>>(a2, K3, a3);
  final_reduce<<<2560, 256, 0, stream>>>(a3, out);
}
